// AttackHead_71657234367194
// MI455X (gfx1250) — compile-verified, matrix operand order checked by reading, not run on hardware
//
#include <hip/hip_runtime.h>

#define TBn   4096
#define Nn    100
#define En    48
#define DLSTM 512
#define DHID  128
#define DKEY  32
#define NEGV  (-1.0e9f)

typedef __attribute__((ext_vector_type(16))) __bf16 v16bf;
typedef __attribute__((ext_vector_type(8)))  __bf16 v8bf;
typedef __attribute__((ext_vector_type(8)))  float  v8f;

// ---------------------------------------------------------------------------
// Wave32 WMMA 16x16x32 bf16 fragment layout (ISA 7.12.2):
//   A (16xK): lane<16 : M=lane,    elems 0..7 = K 0..7,  8..15 = K 16..23
//             lane>=16: M=lane-16, elems 0..7 = K 8..15, 8..15 = K 24..31
//   B (Kx16): identical with N in place of M.
// We store ALL matrix operands pre-swizzled: fragment entry index
//   e = (kchunk*NT + ntile)*32 + lane, each entry = 16 contiguous bf16 (32B)
// so per-lane operand loads are two B128 transactions.
// Mapping kin (0..31) -> (hi, j): kin<16: hi=kin>>3, j=kin&7
//                                 kin>=16: hi=(kin>>3)&1, j=8+(kin&7)
// ---------------------------------------------------------------------------
static __device__ inline void frag_store16(v16bf* buf, int r, int k, __bf16 v) {
  const int kin = k & 31, kc = k >> 5;
  const int hi = (kin < 16) ? (kin >> 3) : ((kin >> 3) & 1);
  const int j  = (kin < 16) ? (kin & 7) : (8 + (kin & 7));
  ((__bf16*)&buf[kc * 32 + hi * 16 + r])[j] = v;
}

// ---------------- weight fp32 [K][128] -> swizzled bf16 fragments ----------------
__global__ __launch_bounds__(256) void k_swz(const float* __restrict__ src,
                                             v16bf* __restrict__ dst,
                                             int Kpad, int Ksrc) {
  const int e = blockIdx.x * 256 + threadIdx.x;  // one fragment-lane (32B) per thread
  const int total = (Kpad >> 5) * 8 * 32;
  if (e >= total) return;
  const int lane = e & 31, nt = (e >> 5) & 7, kc = e >> 8;
  const int n = lane & 15, hi = lane >> 4, col = nt * 16 + n;
  v16bf h;
#pragma unroll
  for (int j = 0; j < 16; ++j) {
    int kin = (j < 8) ? (hi * 8 + j) : (16 + hi * 8 + (j - 8));
    int k = kc * 32 + kin;
    h[j] = (k < Ksrc) ? (__bf16)src[(size_t)k * DHID + col] : (__bf16)0.f;
  }
  dst[e] = h;
}

// ---------------- LayerNorm over 128 cols of a 16x128 tile ----------------
// buf: row-major f32 (stats + values); bufW: swizzled bf16 fragment mirror.
static __device__ inline void ln_inplace(float (&buf)[16][DHID], v16bf* bufW,
                                         float (&redS)[16][16], float (&redQ)[16][16],
                                         const float* __restrict__ g,
                                         const float* __restrict__ be,
                                         bool relu_after, int tid) {
  const int row = tid >> 4, part = tid & 15;
  float s = 0.f, q = 0.f;
#pragma unroll
  for (int j = 0; j < 8; ++j) { float v = buf[row][part * 8 + j]; s += v; q += v * v; }
  redS[row][part] = s; redQ[row][part] = q;
  __syncthreads();
  float ms = 0.f, mq = 0.f;
#pragma unroll
  for (int j = 0; j < 16; ++j) { ms += redS[row][j]; mq += redQ[row][j]; }
  const float mean = ms * (1.f / 128.f);
  const float var  = mq * (1.f / 128.f) - mean * mean;
  const float inv  = rsqrtf(var + 1e-5f);
#pragma unroll
  for (int j = 0; j < 8; ++j) {
    int c = part * 8 + j;
    float v = (buf[row][c] - mean) * inv * g[c] + be[c];
    if (relu_after) v = fmaxf(v, 0.f);
    buf[row][c] = v;
    frag_store16(bufW, row, c, (__bf16)v);
  }
  __syncthreads();
}

// One residual FC block: h=relu(LN(x@w1+b1)); h2=LN(h@w2+b2); x=relu(h2+x)
static __device__ inline void res_block(float (&xF)[16][DHID], v16bf* xw,
                                        float (&hF)[16][DHID], v16bf* hw,
                                        float (&redS)[16][16], float (&redQ)[16][16],
                                        const v16bf* __restrict__ w1, const float* __restrict__ b1,
                                        const float* __restrict__ g1, const float* __restrict__ e1,
                                        const v16bf* __restrict__ w2, const float* __restrict__ b2,
                                        const float* __restrict__ g2, const float* __restrict__ e2,
                                        int tid, int lane, int wv) {
  const int n = lane & 15, mb = (lane >> 4) * 8, col = wv * 16 + n;
  {
    v8f acc = {};
#pragma unroll
    for (int kc = 0; kc < 4; ++kc) {
      v16bf a = xw[kc * 32 + lane];
      v16bf b = w1[(kc * 8 + wv) * 32 + lane];
      acc = __builtin_amdgcn_wmma_f32_16x16x32_bf16(false, a, false, b, (short)0, acc, false, false);
    }
    const float bias = b1[col];
#pragma unroll
    for (int v = 0; v < 8; ++v) hF[mb + v][col] = acc[v] + bias;
  }
  __syncthreads();
  ln_inplace(hF, hw, redS, redQ, g1, e1, true, tid);
  {
    v8f acc2 = {};
#pragma unroll
    for (int kc = 0; kc < 4; ++kc) {
      v16bf a = hw[kc * 32 + lane];
      v16bf b = w2[(kc * 8 + wv) * 32 + lane];
      acc2 = __builtin_amdgcn_wmma_f32_16x16x32_bf16(false, a, false, b, (short)0, acc2, false, false);
    }
    __syncthreads();  // all waves done reading hw before LN rewrites it
    const float bias = b2[col];
#pragma unroll
    for (int v = 0; v < 8; ++v) hF[mb + v][col] = acc2[v] + bias;
  }
  __syncthreads();
  ln_inplace(hF, hw, redS, redQ, g2, e2, false, tid);
  for (int i = tid; i < 16 * DHID; i += 256) {
    int r = i >> 7, c = i & 127;
    float v = fmaxf(hF[r][c] + xF[r][c], 0.f);
    xF[r][c] = v;
    frag_store16(xw, r, c, (__bf16)v);
  }
  __syncthreads();
}

// ---------------- Kernel 1: x = resblocks(relu(lstm @ w_proj + b)) ----------------
__global__ __launch_bounds__(256) void k_xpath(
    const float* __restrict__ lstm,
    const v16bf* __restrict__ wP,  const float* __restrict__ bP,
    const v16bf* __restrict__ w11, const float* __restrict__ b11,
    const float* __restrict__ g11, const float* __restrict__ e11,
    const v16bf* __restrict__ w12, const float* __restrict__ b12,
    const float* __restrict__ g12, const float* __restrict__ e12,
    const v16bf* __restrict__ w21, const float* __restrict__ b21,
    const float* __restrict__ g21, const float* __restrict__ e21,
    const v16bf* __restrict__ w22, const float* __restrict__ b22,
    const float* __restrict__ g22, const float* __restrict__ e22,
    float* __restrict__ xout) {
  __shared__ v16bf lsw[512];        // 16 kchunks * 32 lanes, swizzled lstm tile (16KB)
  __shared__ v16bf xw[128];         // 4 kchunks * 32 lanes (4KB)
  __shared__ v16bf hw[128];
  __shared__ float xF[16][DHID];
  __shared__ float hF[16][DHID];
  __shared__ float redS[16][16];
  __shared__ float redQ[16][16];

  const int tid  = threadIdx.x;
  const int lane = tid & 31;
  const int wv   = tid >> 5;        // wave id = N-tile id
  const int row0 = blockIdx.x * 16;

  // Stage lstm tile [16][512] into swizzled bf16 fragments: 8 consecutive K
  // values for one row land in one lane's contiguous half-fragment (16B store).
  for (int i = tid; i < 1024; i += 256) {
    const int r = i >> 6, c8 = i & 63;            // c8: group of 8 K values
    const int kc = c8 >> 2, sub = c8 & 3;
    const int hi = sub & 1, j0 = (sub >> 1) * 8;
    const float4* src = (const float4*)(lstm + (size_t)(row0 + r) * DLSTM + c8 * 8);
    float4 f0 = src[0], f1 = src[1];
    v8bf h;
    h[0] = (__bf16)f0.x; h[1] = (__bf16)f0.y; h[2] = (__bf16)f0.z; h[3] = (__bf16)f0.w;
    h[4] = (__bf16)f1.x; h[5] = (__bf16)f1.y; h[6] = (__bf16)f1.z; h[7] = (__bf16)f1.w;
    *(v8bf*)((__bf16*)&lsw[kc * 32 + hi * 16 + r] + j0) = h;
  }
  __syncthreads();

  v8f acc = {};
#pragma unroll 4
  for (int kc = 0; kc < 16; ++kc) {
    __builtin_prefetch((const __bf16*)(wP + ((kc + 1) * 8 + wv) * 32), 0, 1);
    v16bf a = lsw[kc * 32 + lane];
    v16bf b = wP[(kc * 8 + wv) * 32 + lane];
    acc = __builtin_amdgcn_wmma_f32_16x16x32_bf16(false, a, false, b, (short)0, acc, false, false);
  }
  {
    const int n = lane & 15, mb = (lane >> 4) * 8, col = wv * 16 + n;
    const float bias = bP[col];
#pragma unroll
    for (int v = 0; v < 8; ++v) {
      float val = fmaxf(acc[v] + bias, 0.f);
      xF[mb + v][col] = val;
      frag_store16(xw, mb + v, col, (__bf16)val);
    }
  }
  __syncthreads();

  res_block(xF, xw, hF, hw, redS, redQ, w11, b11, g11, e11, w12, b12, g12, e12, tid, lane, wv);
  res_block(xF, xw, hF, hw, redS, redQ, w21, b21, g21, e21, w22, b22, g22, e22, tid, lane, wv);

  for (int i = tid; i < 16 * DHID; i += 256) {
    int r = i >> 7, c = i & 127;
    xout[(size_t)(row0 + r) * DHID + c] = xF[r][c];
  }
}

// ---------------- Kernel 2: fused entity GEMM + 3 masked max-pools ----------------
__global__ __launch_bounds__(256) void k_ent_pool(
    const float* __restrict__ ent,  // [TB][N][E]
    const float* __restrict__ mel, const float* __restrict__ rngm,
    const float* __restrict__ mag,
    const v16bf* __restrict__ wEnt,  // swizzled [2 kchunks][8 ntiles][32 lanes]
    const float* __restrict__ bEnt,
    float* __restrict__ pools)       // [TB][3][128]
{
  __shared__ v16bf eAw[64];          // 2 kchunks * 32 lanes (2KB)
  __shared__ float msk[3][112];

  const int tid  = threadIdx.x;
  const int lane = tid & 31;
  const int wv   = tid >> 5;
  const int row  = blockIdx.x;

  for (int i = tid; i < 112; i += 256) {
    float m0 = 0.f, m1 = 0.f, m2 = 0.f;
    if (i < Nn) {
      m0 = mel[(size_t)row * Nn + i];
      m1 = rngm[(size_t)row * Nn + i];
      m2 = mag[(size_t)row * Nn + i];
    }
    msk[0][i] = m0; msk[1][i] = m1; msk[2][i] = m2;
  }

  const int n_ = lane & 15, mb = (lane >> 4) * 8, col = wv * 16 + n_;
  const float bias = bEnt[col];
  float mx0 = -3.0e38f, mx1 = -3.0e38f, mx2 = -3.0e38f;

  for (int t = 0; t < 7; ++t) {
    // Stage A tile [16][48->64] swizzled; groups of 8 E-values -> 16B store.
    for (int i = tid; i < 128; i += 256) {
      const int r = i >> 3, c8 = i & 7;
      const int kc = c8 >> 2, sub = c8 & 3;
      const int hi = sub & 1, j0 = (sub >> 1) * 8;
      const int nn = t * 16 + r;
      v8bf h = {};
      if (c8 < 6 && nn < Nn) {  // En = 48 = 6 groups of 8
        const float4* src = (const float4*)(ent + (size_t)row * Nn * En + (size_t)nn * En + c8 * 8);
        float4 f0 = src[0], f1 = src[1];
        h[0] = (__bf16)f0.x; h[1] = (__bf16)f0.y; h[2] = (__bf16)f0.z; h[3] = (__bf16)f0.w;
        h[4] = (__bf16)f1.x; h[5] = (__bf16)f1.y; h[6] = (__bf16)f1.z; h[7] = (__bf16)f1.w;
      }
      *(v8bf*)((__bf16*)&eAw[kc * 32 + hi * 16 + r] + j0) = h;
    }
    __syncthreads();
    v8f acc = {};
#pragma unroll
    for (int kc = 0; kc < 2; ++kc) {
      v16bf a = eAw[kc * 32 + lane];
      v16bf b = wEnt[(kc * 8 + wv) * 32 + lane];
      acc = __builtin_amdgcn_wmma_f32_16x16x32_bf16(false, a, false, b, (short)0, acc, false, false);
    }
#pragma unroll
    for (int v = 0; v < 8; ++v) {
      int nn = t * 16 + mb + v;
      if (nn < Nn) {
        float val = acc[v] + bias;
        mx0 = fmaxf(mx0, val * msk[0][nn]);
        mx1 = fmaxf(mx1, val * msk[1][nn]);
        mx2 = fmaxf(mx2, val * msk[2][nn]);
      }
    }
    __syncthreads();
  }
  mx0 = fmaxf(mx0, __shfl_xor(mx0, 16));
  mx1 = fmaxf(mx1, __shfl_xor(mx1, 16));
  mx2 = fmaxf(mx2, __shfl_xor(mx2, 16));
  if (lane < 16) {
    float* p = pools + (size_t)row * 3 * DHID;
    p[0 * DHID + col] = mx0;
    p[1 * DHID + col] = mx1;
    p[2 * DHID + col] = mx2;
  }
}

// ---------------- Kernel 3: type logits, argmax action, q vector ----------------
__global__ __launch_bounds__(128) void k_type_q(
    const float* __restrict__ lstm, const float* __restrict__ x,
    const float* __restrict__ pools, const float* __restrict__ notatk,
    const float* __restrict__ isatk,
    const float* __restrict__ wq1, const float* __restrict__ bq1,
    const float* __restrict__ wq2, const float* __restrict__ bq2,
    float* __restrict__ out_lt, float* __restrict__ out_act_f,
    int* __restrict__ act_i, float* __restrict__ qout) {
  __shared__ float ls[DLSTM];
  __shared__ float red[4][128];
  __shared__ float t1[DKEY];

  const int tid = threadIdx.x;
  const int row = blockIdx.x;

  for (int i = tid; i < DLSTM; i += 128) ls[i] = lstm[(size_t)row * DLSTM + i];

  const float xv = x[(size_t)row * DHID + tid];
  const float* prow = pools + (size_t)row * 3 * DHID;
  red[0][tid] = xv * notatk[tid];
  red[1][tid] = xv * prow[0 * DHID + tid];
  red[2][tid] = xv * prow[1 * DHID + tid];
  red[3][tid] = xv * prow[2 * DHID + tid];
  __syncthreads();
  for (int s = 64; s > 0; s >>= 1) {
    if (tid < s) {
#pragma unroll
      for (int a = 0; a < 4; ++a) red[a][tid] += red[a][tid + s];
    }
    __syncthreads();
  }
  if (tid == 0) {
    float best = -3.4e38f; int bi = 0;
#pragma unroll
    for (int a = 0; a < 4; ++a) {
      float lv = (isatk[(size_t)row * 4 + a] != 0.f) ? red[a][0] : NEGV;
      out_lt[(size_t)row * 4 + a] = lv;
      if (lv > best) { best = lv; bi = a; }
    }
    act_i[row] = bi;
    out_act_f[row] = (float)bi;
  }
  __syncthreads();

  {
    const int j = tid & 31, part = tid >> 5;
    float s = 0.f;
    const int k0 = part * 128;
    for (int k = k0; k < k0 + 128; ++k) s += ls[k] * wq1[(size_t)k * DKEY + j];
    red[part][j] = s;
  }
  __syncthreads();
  if (tid < DKEY) {
    float s = red[0][tid] + red[1][tid] + red[2][tid] + red[3][tid] + bq1[tid];
    t1[tid] = fmaxf(s, 0.f);
  }
  __syncthreads();
  if (tid < DKEY) {
    float s = bq2[tid];
    for (int k = 0; k < DKEY; ++k) s += t1[k] * wq2[(size_t)k * DKEY + tid];
    qout[(size_t)row * DKEY + tid] = s;
  }
}

// ---------------- Kernel 4: target logits via qk = w_key @ q reduction ----------------
__global__ __launch_bounds__(128) void k_target(
    const float* __restrict__ ent, const float* __restrict__ mel,
    const float* __restrict__ rngm, const float* __restrict__ mag,
    const float* __restrict__ wkey, const float* __restrict__ bkey,
    const float* __restrict__ q, const int* __restrict__ action,
    float* __restrict__ out_tg) {
  __shared__ float qs[DKEY];
  __shared__ float qk[En];
  __shared__ float qb_s;

  const int tid = threadIdx.x;
  const int row = blockIdx.x;
  if (tid < DKEY) qs[tid] = q[(size_t)row * DKEY + tid];
  __syncthreads();
  if (tid < En) {
    float s = 0.f;
    for (int d = 0; d < DKEY; ++d) s += wkey[(size_t)tid * DKEY + d] * qs[d];
    qk[tid] = s;
  }
  if (tid == 0) {
    float s = 0.f;
    for (int d = 0; d < DKEY; ++d) s += bkey[d] * qs[d];
    qb_s = s;
  }
  __syncthreads();
  const int act = action[row];
  if (tid < Nn) {
    float sel;
    if (act == 0)      sel = 1.f;
    else if (act == 1) sel = mel[(size_t)row * Nn + tid];
    else if (act == 2) sel = rngm[(size_t)row * Nn + tid];
    else               sel = mag[(size_t)row * Nn + tid];
    float lv = NEGV;
    if (sel != 0.f) {
      float s = qb_s;
      const float4* e4 = (const float4*)(ent + (size_t)row * Nn * En + (size_t)tid * En);
#pragma unroll
      for (int k4 = 0; k4 < En / 4; ++k4) {
        float4 f = e4[k4];
        s += f.x * qk[4 * k4] + f.y * qk[4 * k4 + 1] + f.z * qk[4 * k4 + 2] + f.w * qk[4 * k4 + 3];
      }
      lv = s;
    }
    out_tg[(size_t)row * Nn + tid] = lv;
  }
}

// ---------------- host launch ----------------
extern "C" void kernel_launch(void* const* d_in, const int* in_sizes, int n_in,
                              void* d_out, int out_size, void* d_ws, size_t ws_size,
                              hipStream_t stream) {
  (void)in_sizes; (void)n_in; (void)out_size; (void)ws_size;
  const float* lstm  = (const float*)d_in[0];
  const float* rngm  = (const float*)d_in[1];
  const float* mel   = (const float*)d_in[2];
  const float* mag   = (const float*)d_in[3];
  const float* ent   = (const float*)d_in[4];
  const float* isatk = (const float*)d_in[6];
  const float* w_proj = (const float*)d_in[7];  const float* b_proj = (const float*)d_in[8];
  const float* r1w1 = (const float*)d_in[9];   const float* r1b1 = (const float*)d_in[10];
  const float* r1g1 = (const float*)d_in[11];  const float* r1e1 = (const float*)d_in[12];
  const float* r1w2 = (const float*)d_in[13];  const float* r1b2 = (const float*)d_in[14];
  const float* r1g2 = (const float*)d_in[15];  const float* r1e2 = (const float*)d_in[16];
  const float* r2w1 = (const float*)d_in[17];  const float* r2b1 = (const float*)d_in[18];
  const float* r2g1 = (const float*)d_in[19];  const float* r2e1 = (const float*)d_in[20];
  const float* r2w2 = (const float*)d_in[21];  const float* r2b2 = (const float*)d_in[22];
  const float* r2g2 = (const float*)d_in[23];  const float* r2e2 = (const float*)d_in[24];
  const float* w_ent = (const float*)d_in[25]; const float* b_ent = (const float*)d_in[26];
  const float* notatk = (const float*)d_in[27];
  const float* w_key = (const float*)d_in[28]; const float* b_key = (const float*)d_in[29];
  const float* w_q1 = (const float*)d_in[30];  const float* b_q1 = (const float*)d_in[31];
  const float* w_q2 = (const float*)d_in[32];  const float* b_q2 = (const float*)d_in[33];

  char* ws = (char*)d_ws;
  v16bf* wbP  = (v16bf*)(ws + 0);        // 512*128*2 B
  v16bf* wb11 = (v16bf*)(ws + 131072);   // 128*128*2 B each
  v16bf* wb12 = (v16bf*)(ws + 163840);
  v16bf* wb21 = (v16bf*)(ws + 196608);
  v16bf* wb22 = (v16bf*)(ws + 229376);
  v16bf* wbE  = (v16bf*)(ws + 262144);   // 64*128*2 B
  float* xbuf = (float*)(ws + 278528);   // 4096*128*4
  float* pools= (float*)(ws + 2375680);  // 4096*3*128*4
  float* qbuf = (float*)(ws + 8667136);  // 4096*32*4
  int*   act  = (int*)(ws + 9191424);    // 4096*4

  k_swz<<<16, 256, 0, stream>>>(w_proj, wbP, 512, 512);
  k_swz<<<4, 256, 0, stream>>>(r1w1, wb11, 128, 128);
  k_swz<<<4, 256, 0, stream>>>(r1w2, wb12, 128, 128);
  k_swz<<<4, 256, 0, stream>>>(r2w1, wb21, 128, 128);
  k_swz<<<4, 256, 0, stream>>>(r2w2, wb22, 128, 128);
  k_swz<<<2, 256, 0, stream>>>(w_ent, wbE, 64, 48);

  k_xpath<<<TBn / 16, 256, 0, stream>>>(lstm, wbP, b_proj,
      wb11, r1b1, r1g1, r1e1, wb12, r1b2, r1g2, r1e2,
      wb21, r2b1, r2g1, r2e1, wb22, r2b2, r2g2, r2e2, xbuf);

  k_ent_pool<<<TBn, 256, 0, stream>>>(ent, mel, rngm, mag, wbE, b_ent, pools);

  float* out_lt  = (float*)d_out;
  float* out_act = out_lt + (size_t)TBn * 4;
  float* out_tg  = out_act + TBn;

  k_type_q<<<TBn, 128, 0, stream>>>(lstm, xbuf, pools, notatk, isatk,
      w_q1, b_q1, w_q2, b_q2, out_lt, out_act, act, qbuf);

  k_target<<<TBn, 128, 0, stream>>>(ent, mel, rngm, mag, w_key, b_key, qbuf, act, out_tg);
}
